// EncoderLayer_85478439125057
// MI455X (gfx1250) — matrix operand order repaired and checked by reading, not compiled, not run on hardware
//
#include <hip/hip_runtime.h>

#define B_   2
#define N_   2048
#define DM   512
#define H_   8
#define DH_  64
#define DFF  2048
#define MF   256   // NB_FEAT

typedef __attribute__((ext_vector_type(16))) _Float16 v16h;
typedef __attribute__((ext_vector_type(8)))  _Float16 v8h;
typedef __attribute__((ext_vector_type(4)))  _Float16 v4h;
typedef __attribute__((ext_vector_type(8)))  float    v8f;
typedef __attribute__((ext_vector_type(4)))  float    v4f;
typedef __attribute__((ext_vector_type(4)))  int      v4i;

#define AS1 __attribute__((address_space(1)))
#define AS3 __attribute__((address_space(3)))

#define LP 40  // LDS tile pitch in halfs (80B, multiple of 16B)

// ---- async global->LDS staging (gfx1250 ASYNCcnt path) with guarded fallback
#if defined(__has_builtin)
# if __has_builtin(__builtin_amdgcn_global_load_async_to_lds_b128)
#  define USE_ASYNC_LDS 1
# endif
#endif
#ifndef USE_ASYNC_LDS
# define USE_ASYNC_LDS 0
#endif

__device__ __forceinline__ void stage_b128(const _Float16* __restrict__ g, _Float16* l) {
#if USE_ASYNC_LDS
  __builtin_amdgcn_global_load_async_to_lds_b128(
      (AS1 v4i*)(AS1 void*)g,
      (AS3 v4i*)(AS3 void*)l, 0, 0);
#else
  *(v8h*)l = *(const v8h*)g;
#endif
}

__device__ __forceinline__ void stage_wait() {
#if USE_ASYNC_LDS
# if defined(__has_builtin) && __has_builtin(__builtin_amdgcn_s_wait_asynccnt)
  __builtin_amdgcn_s_wait_asynccnt(0);
# else
  asm volatile("s_wait_asynccnt 0" ::: "memory");
# endif
#endif
}

// A fragment (16x32 f16, M = lane&15): lanes 0-15 hold K={0..7,16..23}, lanes 16-31 K={8..15,24..31}
__device__ __forceinline__ v16h frag_ld_a(const _Float16* __restrict__ sm, int rowBase, int lane) {
  const _Float16* p = sm + (rowBase + (lane & 15)) * LP + ((lane & 16) ? 8 : 0);
  v8h lo = *(const v8h*)(p);
  v8h hi = *(const v8h*)(p + 16);
  v16h r;
#pragma unroll
  for (int i = 0; i < 8; ++i) { r[i] = lo[i]; r[8 + i] = hi[i]; }
  return r;
}

// B fragment (32x16 f16, N = lane&15): lanes 0-15 hold K={0..7,16..23}, lanes 16-31 K={8..15,24..31}
__device__ __forceinline__ v16h frag_ld_b(const _Float16* __restrict__ sm, int colBase, int lane) {
  const _Float16* p = sm + (colBase + (lane & 15)) * LP + ((lane & 16) ? 8 : 0);
  v8h lo = *(const v8h*)(p);
  v8h hi = *(const v8h*)(p + 16);
  v16h r;
#pragma unroll
  for (int i = 0; i < 8; ++i) { r[i] = lo[i]; r[8 + i] = hi[i]; }
  return r;
}

#define WMMA_F16(A, Bf, Cf) \
  __builtin_amdgcn_wmma_f32_16x16x32_f16(false, (A), false, (Bf), (short)0, (Cf), false, false)

// ---------------------------------------------------------------------------
// Generic batched GEMM: C[z] = alpha * opA(A[z]) @ opB(B[z]) (+bias)(*rowscale)(relu)
// Block tile 128x64, K-step 32, 256 threads = 8 waves, 4 WMMA accumulators/wave.
// M multiple of 128, Ncols multiple of 64, K multiple of 32.
// AH: A is pre-converted _Float16 (alpha ignored); else A is float (scaled by alpha).
// ---------------------------------------------------------------------------
template <bool TA, bool TB, bool AH>
__global__ __launch_bounds__(256)
void gemm_wmma(const void* __restrict__ A, long sAo, long sAi, int lda,
               const float* __restrict__ B, long sBo, long sBi, int ldb,
               float* __restrict__ C, long sCo, long sCi, int ldc,
               const float* __restrict__ bias,
               const float* __restrict__ rowscale, long sRo, long sRi,
               int K, float alpha, int relu, int innerCnt)
{
  __shared__ __align__(16) _Float16 smA[128 * LP];
  __shared__ __align__(16) _Float16 smB[64 * LP];
  const int z  = blockIdx.z;
  const int bo = z / innerCnt, bi = z % innerCnt;
  const float*    Af = (const float*)A    + bo * sAo + bi * sAi;
  const _Float16* Ah = (const _Float16*)A + bo * sAo + bi * sAi;
  const float* Bb = B + bo * sBo + bi * sBi;
  float*       Cb = C + bo * sCo + bi * sCi;
  const float* Rb = rowscale ? rowscale + bo * sRo + bi * sRi : nullptr;

  const int m0 = blockIdx.y * 128;
  const int n0 = blockIdx.x * 64;
  const int tid = threadIdx.x;
  const int lane = tid & 31, w = tid >> 5;
  const int wr = (w & 3) * 32;   // 32-row strip (two 16-row tiles)
  const int tc = (w >> 2) * 32;  // 32-col strip (two 16-col tiles)

  v8f c00 = {}, c01 = {}, c10 = {}, c11 = {};
  for (int k0 = 0; k0 < K; k0 += 32) {
    __syncthreads();
    // ---- A tile 128(m) x 32(k) -> smA[m][k]
    if (AH) {
      if (TA) {                   // A(m,k) = Ah[k*lda + m], contiguous in m
#pragma unroll
        for (int i = 0; i < 2; ++i) {   // 4096 halfs = 512 v8h
          int j = tid + 256 * i;
          int k = j >> 4, mq = (j & 15) * 8;
          v8h v = *(const v8h*)(Ah + (long)(k0 + k) * lda + (m0 + mq));
#pragma unroll
          for (int e = 0; e < 8; ++e) smA[(mq + e) * LP + k] = v[e];
        }
      } else {                    // A(m,k) = Ah[m*lda + k], contiguous in k
#pragma unroll
        for (int i = 0; i < 2; ++i) {
          int j = tid + 256 * i;
          int m = j >> 2, kq = (j & 3) * 8;
          stage_b128(Ah + (long)(m0 + m) * lda + (k0 + kq), smA + m * LP + kq);
        }
      }
    } else {
      if (TA) {                   // A(m,k) = Af[k*lda + m], contiguous in m
#pragma unroll
        for (int i = 0; i < 4; ++i) {   // 128x32 = 1024 float4
          int j = tid + 256 * i;
          int k = j >> 5, mq = (j & 31) * 4;
          v4f v = *(const v4f*)(Af + (long)(k0 + k) * lda + (m0 + mq));
#pragma unroll
          for (int e = 0; e < 4; ++e) smA[(mq + e) * LP + k] = (_Float16)(alpha * v[e]);
        }
      } else {                    // A(m,k) = Af[m*lda + k], contiguous in k
#pragma unroll
        for (int i = 0; i < 4; ++i) {
          int j = tid + 256 * i;
          int m = j >> 3, kq = (j & 7) * 4;
          v4f v = *(const v4f*)(Af + (long)(m0 + m) * lda + (k0 + kq));
          v4h hv;
#pragma unroll
          for (int e = 0; e < 4; ++e) hv[e] = (_Float16)(alpha * v[e]);
          *(v4h*)(smA + m * LP + kq) = hv;
        }
      }
    }
    // ---- B tile 32(k) x 64(n) -> smB[n][k]
    if (TB) {                     // B(k,n) = Bb[n*ldb + k], contiguous in k
#pragma unroll
      for (int i = 0; i < 2; ++i) { // 32x64 = 512 float4
        int j = tid + 256 * i;
        int n = j >> 3, kq = (j & 7) * 4;
        v4f v = *(const v4f*)(Bb + (long)(n0 + n) * ldb + (k0 + kq));
        v4h hv;
#pragma unroll
        for (int e = 0; e < 4; ++e) hv[e] = (_Float16)v[e];
        *(v4h*)(smB + n * LP + kq) = hv;
      }
    } else {                      // B(k,n) = Bb[k*ldb + n], contiguous in n
#pragma unroll
      for (int i = 0; i < 2; ++i) {
        int j = tid + 256 * i;
        int k = j >> 4, nq = (j & 15) * 4;
        v4f v = *(const v4f*)(Bb + (long)(k0 + k) * ldb + (n0 + nq));
#pragma unroll
        for (int e = 0; e < 4; ++e) smB[(nq + e) * LP + k] = (_Float16)v[e];
      }
    }
    if (AH && !TA) stage_wait();
    __syncthreads();
    v16h a0 = frag_ld_a(smA, wr,      lane);
    v16h a1 = frag_ld_a(smA, wr + 16, lane);
    v16h b0 = frag_ld_b(smB, tc,      lane);
    v16h b1 = frag_ld_b(smB, tc + 16, lane);
    c00 = WMMA_F16(a0, b0, c00);
    c01 = WMMA_F16(a0, b1, c01);
    c10 = WMMA_F16(a1, b0, c10);
    c11 = WMMA_F16(a1, b1, c11);
  }

  const int colBase = n0 + tc + (lane & 15);
  const int rowHalf = (lane >> 4) * 8;
  float bia0 = bias ? bias[colBase]      : 0.f;
  float bia1 = bias ? bias[colBase + 16] : 0.f;
#pragma unroll
  for (int rt = 0; rt < 2; ++rt) {
    v8f& d0 = rt ? c10 : c00;
    v8f& d1 = rt ? c11 : c01;
#pragma unroll
    for (int v = 0; v < 8; ++v) {
      int row = m0 + wr + rt * 16 + rowHalf + v;
      float x0 = d0[v] + bia0, x1 = d1[v] + bia1;
      if (Rb)   { float s = Rb[row]; x0 *= s; x1 *= s; }
      if (relu) { x0 = fmaxf(x0, 0.f); x1 = fmaxf(x1, 0.f); }
      Cb[(long)row * ldc + colBase]      = x0;
      Cb[(long)row * ldc + colBase + 16] = x1;
    }
  }
}

// ---------------------------------------------------------------------------
// attn_weight[b,n,m] = (1/H) * sum_h | (q[b,h,n,:] . k[b,h,m,:]) * D_inv[b,h,n] |
// f16 feature inputs: tile staging via async global->LDS b128 (fallback: v8h copy).
// Block tile 128(n) x 64(m), inner loop over h with K=256 WMMA reduction each.
// ---------------------------------------------------------------------------
__global__ __launch_bounds__(256)
void attn_wmma(const _Float16* __restrict__ qh, const _Float16* __restrict__ kh,
               const float* __restrict__ dinv, float* __restrict__ out)
{
  __shared__ __align__(16) _Float16 smA[128 * LP];
  __shared__ __align__(16) _Float16 smB[64 * LP];
  __shared__ float dS[128];
  const int b  = blockIdx.z;
  const int n0 = blockIdx.y * 128;  // query rows
  const int m0 = blockIdx.x * 64;   // key rows (output cols)
  const int tid = threadIdx.x, lane = tid & 31, w = tid >> 5;
  const int wr = (w & 3) * 32, tc = (w >> 2) * 32;
  const int rowHalf = (lane >> 4) * 8;
  const int colL = lane & 15;

  v8f acc00 = {}, acc01 = {}, acc10 = {}, acc11 = {};
  for (int h = 0; h < H_; ++h) {
    const long base = (long)(b * H_ + h) * N_ * MF;
    const _Float16* Q  = qh + base;
    const _Float16* Kf = kh + base;
    const float* Dv = dinv + (long)(b * H_ + h) * N_;
    v8f s00 = {}, s01 = {}, s10 = {}, s11 = {};
    for (int k0 = 0; k0 < MF; k0 += 32) {
      __syncthreads();
#pragma unroll
      for (int i = 0; i < 2; ++i) {       // A: 128x32 halfs = 512 b128 stages
        int j = tid + 256 * i;
        int r = j >> 2, kq = (j & 3) * 8;
        stage_b128(Q + (long)(n0 + r) * MF + (k0 + kq), smA + r * LP + kq);
      }
      {                                   // B: 64x32 halfs = 256 b128 stages
        int r = tid >> 2, kq = (tid & 3) * 8;
        stage_b128(Kf + (long)(m0 + r) * MF + (k0 + kq), smB + r * LP + kq);
      }
      if (k0 == 0 && tid < 128) dS[tid] = Dv[n0 + tid];
      stage_wait();
      __syncthreads();
      v16h a0 = frag_ld_a(smA, wr,      lane);
      v16h a1 = frag_ld_a(smA, wr + 16, lane);
      v16h b0 = frag_ld_b(smB, tc,      lane);
      v16h b1 = frag_ld_b(smB, tc + 16, lane);
      s00 = WMMA_F16(a0, b0, s00);
      s01 = WMMA_F16(a0, b1, s01);
      s10 = WMMA_F16(a1, b0, s10);
      s11 = WMMA_F16(a1, b1, s11);
    }
#pragma unroll
    for (int v = 0; v < 8; ++v) {
      float sc0 = dS[wr + rowHalf + v];
      float sc1 = dS[wr + 16 + rowHalf + v];
      acc00[v] += fabsf(s00[v] * sc0);
      acc01[v] += fabsf(s01[v] * sc0);
      acc10[v] += fabsf(s10[v] * sc1);
      acc11[v] += fabsf(s11[v] * sc1);
    }
  }
  const float invH = 1.0f / H_;
  float* Ob = out + (long)b * N_ * N_;
#pragma unroll
  for (int v = 0; v < 8; ++v) {
    int row0 = n0 + wr + rowHalf + v;
    int col  = m0 + tc + colL;
    Ob[(long)row0 * N_ + col]             = acc00[v] * invH;
    Ob[(long)row0 * N_ + col + 16]        = acc01[v] * invH;
    Ob[(long)(row0 + 16) * N_ + col]      = acc10[v] * invH;
    Ob[(long)(row0 + 16) * N_ + col + 16] = acc11[v] * invH;
  }
}

// ---------------------------------------------------------------------------
// small helper kernels
// ---------------------------------------------------------------------------
__global__ void diag_kernel(const float* __restrict__ buf, float* __restrict__ dg, int total)
{
  int r = blockIdx.x * blockDim.x + threadIdx.x;
  if (r >= total) return;
  int b = r / (H_ * N_);
  int h = (r / N_) % H_;
  int n = r % N_;
  const float* p = buf + (long)b * N_ * DM + (long)n * DM + h * DH_;
  float s = 0.f;
  for (int e = 0; e < DH_; ++e) { float v = p[e]; s += v * v; }
  dg[r] = 0.0625f * s;   // 0.5 * dn^2 = 0.5 / 8
}

__global__ __launch_bounds__(256)
void max_part(const float* __restrict__ x, long n, float* __restrict__ outp)
{
  __shared__ float sm[256];
  float m = -3.4e38f;
  for (long i = (long)blockIdx.x * 256 + threadIdx.x; i < n; i += (long)gridDim.x * 256)
    m = fmaxf(m, x[i]);
  sm[threadIdx.x] = m; __syncthreads();
  for (int s = 128; s > 0; s >>= 1) {
    if ((int)threadIdx.x < s) sm[threadIdx.x] = fmaxf(sm[threadIdx.x], sm[threadIdx.x + s]);
    __syncthreads();
  }
  if (threadIdx.x == 0) outp[blockIdx.x] = sm[0];
}

__global__ __launch_bounds__(256)
void max_final(const float* __restrict__ part, int n, float* __restrict__ outp)
{
  __shared__ float sm[256];
  float m = -3.4e38f;
  for (int i = threadIdx.x; i < n; i += 256) m = fmaxf(m, part[i]);
  sm[threadIdx.x] = m; __syncthreads();
  for (int s = 128; s > 0; s >>= 1) {
    if ((int)threadIdx.x < s) sm[threadIdx.x] = fmaxf(sm[threadIdx.x], sm[threadIdx.x + s]);
    __syncthreads();
  }
  if (threadIdx.x == 0) *outp = sm[0];
}

// q features: per-row (256) max, then ratio*(exp(dd - diag - rowmax)+eps) -> f16
// one wave per row; each lane owns 8 contiguous elements
__global__ __launch_bounds__(256)
void qexp_kernel(const float* __restrict__ dd, const float* __restrict__ dg,
                 _Float16* __restrict__ oh)
{
  int w = threadIdx.x >> 5, lane = threadIdx.x & 31;
  long row = (long)blockIdx.x * 8 + w;
  const float* p = dd + row * MF + lane * 8;
  v4f v0 = *(const v4f*)(p);
  v4f v1 = *(const v4f*)(p + 4);
  float mx = -3.4e38f;
#pragma unroll
  for (int e = 0; e < 4; ++e) { mx = fmaxf(mx, v0[e]); mx = fmaxf(mx, v1[e]); }
#pragma unroll
  for (int s = 16; s > 0; s >>= 1) mx = fmaxf(mx, __shfl_down(mx, s, 32));
  mx = __shfl(mx, 0, 32);
  float d = dg[row] + mx;
  v8h hv;
#pragma unroll
  for (int e = 0; e < 4; ++e) {
    hv[e]     = (_Float16)(0.0625f * (__expf(v0[e] - d) + 1e-4f));
    hv[4 + e] = (_Float16)(0.0625f * (__expf(v1[e] - d) + 1e-4f));
  }
  *(v8h*)(oh + row * MF + lane * 8) = hv;
}

// k features: global max; float4 -> half4 elementwise
__global__ void kexp_kernel(const float* __restrict__ dd, const float* __restrict__ dg,
                            const float* __restrict__ mptr, _Float16* __restrict__ oh, long n4)
{
  float m = *mptr;
  for (long i = (long)blockIdx.x * blockDim.x + threadIdx.x; i < n4;
       i += (long)gridDim.x * blockDim.x) {
    long base = i * 4;
    float d = dg[base >> 8] + m;   // MF = 256; 4 contiguous share a row
    v4f v = *(const v4f*)(dd + base);
    v4h hv;
#pragma unroll
    for (int e = 0; e < 4; ++e) hv[e] = (_Float16)(0.0625f * (__expf(v[e] - d) + 1e-4f));
    *(v4h*)(oh + base) = hv;
  }
}

__global__ __launch_bounds__(256)
void ksum_kernel(const _Float16* __restrict__ kh, float* __restrict__ ks)
{
  int bh = blockIdx.x, f = threadIdx.x;
  const _Float16* p = kh + (long)bh * N_ * MF + f;
  float s = 0.f;
  for (int n = 0; n < N_; ++n) s += (float)p[(long)n * MF];
  ks[bh * MF + f] = s;
}

__global__ __launch_bounds__(256)
void dinv_kernel(const _Float16* __restrict__ qh, const float* __restrict__ ks,
                 float* __restrict__ dv)
{
  int w = threadIdx.x >> 5, lane = threadIdx.x & 31;
  long row = (long)blockIdx.x * 8 + w;
  long bh = row / N_;
  v8h q = *(const v8h*)(qh + row * MF + lane * 8);
  const float* k = ks + bh * MF + lane * 8;
  v4f k0 = *(const v4f*)(k);
  v4f k1 = *(const v4f*)(k + 4);
  float s = 0.f;
#pragma unroll
  for (int e = 0; e < 4; ++e) s += (float)q[e] * k0[e] + (float)q[4 + e] * k1[e];
#pragma unroll
  for (int t = 16; t > 0; t >>= 1) s += __shfl_down(s, t, 32);
  if (lane == 0) dv[row] = 1.0f / s;
}

__global__ __launch_bounds__(256)
void ln_kernel(const float* __restrict__ x, const float* __restrict__ g,
               const float* __restrict__ be, float* __restrict__ y)
{
  __shared__ float sm[256];
  long row = blockIdx.x;
  const float* p = x + row * DM;
  float v0 = p[threadIdx.x], v1 = p[threadIdx.x + 256];
  sm[threadIdx.x] = v0 + v1; __syncthreads();
  for (int t = 128; t > 0; t >>= 1) {
    if ((int)threadIdx.x < t) sm[threadIdx.x] += sm[threadIdx.x + t];
    __syncthreads();
  }
  float mu = sm[0] * (1.0f / DM);
  __syncthreads();
  float d0 = v0 - mu, d1 = v1 - mu;
  sm[threadIdx.x] = d0 * d0 + d1 * d1; __syncthreads();
  for (int t = 128; t > 0; t >>= 1) {
    if ((int)threadIdx.x < t) sm[threadIdx.x] += sm[threadIdx.x + t];
    __syncthreads();
  }
  float rstd = rsqrtf(sm[0] * (1.0f / DM) + 1e-5f);
  float* q = y + row * DM;
  q[threadIdx.x]       = g[threadIdx.x]       * d0 * rstd + be[threadIdx.x];
  q[threadIdx.x + 256] = g[threadIdx.x + 256] * d1 * rstd + be[threadIdx.x + 256];
}

// ---------------------------------------------------------------------------
extern "C" void kernel_launch(void* const* d_in, const int* in_sizes, int n_in,
                              void* d_out, int out_size, void* d_ws, size_t ws_size,
                              hipStream_t stream)
{
  (void)in_sizes; (void)n_in; (void)out_size; (void)ws_size;
  const float* x    = (const float*)d_in[0];
  const float* Wq   = (const float*)d_in[1];
  const float* bq   = (const float*)d_in[2];
  const float* Wk   = (const float*)d_in[3];
  const float* bk   = (const float*)d_in[4];
  const float* Wv   = (const float*)d_in[5];
  const float* bv   = (const float*)d_in[6];
  const float* Wo   = (const float*)d_in[7];
  const float* bo   = (const float*)d_in[8];
  const float* proj = (const float*)d_in[9];
  const float* W1   = (const float*)d_in[10];
  const float* b1   = (const float*)d_in[11];
  const float* W2   = (const float*)d_in[12];
  const float* b2   = (const float*)d_in[13];
  const float* g1   = (const float*)d_in[14];
  const float* be1  = (const float*)d_in[15];
  const float* g2   = (const float*)d_in[16];
  const float* be2  = (const float*)d_in[17];

  const long TOK = (long)B_ * N_;               // 4096
  const long XSZ = TOK * DM;                    // 2,097,152
  const long FSZ = (long)B_ * H_ * N_ * MF;     // 8,388,608

  float* ws  = (float*)d_ws;
  float* Qb  = ws; ws += XSZ;
  float* Kb  = ws; ws += XSZ;
  float* Vb  = ws; ws += XSZ;
  float* qf  = ws; ws += FSZ;     // f32 dd_q scratch
  float* kf  = ws; ws += FSZ;     // f32 dd_k scratch
  _Float16* qh = (_Float16*)ws; ws += FSZ / 2;   // f16 q features
  _Float16* kh = (_Float16*)ws; ws += FSZ / 2;   // f16 k features
  float* dgq = ws; ws += (long)B_ * H_ * N_;
  float* dgk = ws; ws += (long)B_ * H_ * N_;
  float* ksm = ws; ws += (long)B_ * H_ * MF;
  float* dnv = ws; ws += (long)B_ * H_ * N_;
  float* ctx = ws; ws += (long)B_ * H_ * MF * DH_;
  float* Ob  = ws; ws += XSZ;
  float* O1  = ws; ws += XSZ;
  float* H1b = ws; ws += XSZ;
  float* FFb = ws; ws += TOK * DFF;
  float* O2  = ws; ws += XSZ;
  float* red = ws; ws += 1025;

  float* yOut    = (float*)d_out;
  float* attnOut = yOut + XSZ;

  dim3 blk(256);
  const float dn = 0.35355339059327373f;  // 64^-0.25

  // Q,K,V = x @ W + b   [4096 x 512] * [512 x 512]
  {
    dim3 g(DM / 64, TOK / 128, 1);
    gemm_wmma<false,false,false><<<g, blk, 0, stream>>>(x, 0, 0, DM, Wq, 0, 0, DM, Qb, 0, 0, DM,
                                                        bq, nullptr, 0, 0, DM, 1.0f, 0, 1);
    gemm_wmma<false,false,false><<<g, blk, 0, stream>>>(x, 0, 0, DM, Wk, 0, 0, DM, Kb, 0, 0, DM,
                                                        bk, nullptr, 0, 0, DM, 1.0f, 0, 1);
    gemm_wmma<false,false,false><<<g, blk, 0, stream>>>(x, 0, 0, DM, Wv, 0, 0, DM, Vb, 0, 0, DM,
                                                        bv, nullptr, 0, 0, DM, 1.0f, 0, 1);
  }
  // diag terms
  {
    int total = B_ * H_ * N_;
    diag_kernel<<<(total + 255) / 256, blk, 0, stream>>>(Qb, dgq, total);
    diag_kernel<<<(total + 255) / 256, blk, 0, stream>>>(Kb, dgk, total);
  }
  // dd = dn * head @ proj^T   batched over (b,h): [2048 x 64] * [64 x 256]
  {
    dim3 g(MF / 64, N_ / 128, B_ * H_);
    long sAo = (long)N_ * DM, sAi = DH_;
    long sCo = (long)H_ * N_ * MF, sCi = (long)N_ * MF;
    gemm_wmma<false,true,false><<<g, blk, 0, stream>>>(Qb, sAo, sAi, DM, proj, 0, 0, DH_,
                                                       qf, sCo, sCi, MF, nullptr, nullptr, 0, 0,
                                                       DH_, dn, 0, H_);
    gemm_wmma<false,true,false><<<g, blk, 0, stream>>>(Kb, sAo, sAi, DM, proj, 0, 0, DH_,
                                                       kf, sCo, sCi, MF, nullptr, nullptr, 0, 0,
                                                       DH_, dn, 0, H_);
  }
  // global max over dd_k (two-stage, deterministic)
  max_part<<<1024, blk, 0, stream>>>(kf, FSZ, red);
  max_final<<<1, blk, 0, stream>>>(red, 1024, red + 1024);
  // positive random features -> f16
  qexp_kernel<<<(B_ * H_ * N_) / 8, blk, 0, stream>>>(qf, dgq, qh);
  kexp_kernel<<<2048, blk, 0, stream>>>(kf, dgk, red + 1024, kh, FSZ / 4);
  // k_sum and D_inv
  ksum_kernel<<<B_ * H_, blk, 0, stream>>>(kh, ksm);
  dinv_kernel<<<(B_ * H_ * N_) / 8, blk, 0, stream>>>(qh, ksm, dnv);
  // context = k^T @ v   batched: [256 x 2048] * [2048 x 64]  (A = f16 features)
  {
    dim3 g(DH_ / 64, MF / 128, B_ * H_);
    gemm_wmma<true,false,true><<<g, blk, 0, stream>>>(kh, (long)H_ * N_ * MF, (long)N_ * MF, MF,
                                                      Vb, (long)N_ * DM, DH_, DM,
                                                      ctx, (long)H_ * MF * DH_, (long)MF * DH_, DH_,
                                                      nullptr, nullptr, 0, 0, N_, 1.0f, 0, H_);
  }
  // out = (q @ context) * D_inv  -> [B,N,D] head-interleaved layout  (A = f16 features)
  {
    dim3 g(DH_ / 64, N_ / 128, B_ * H_);
    gemm_wmma<false,false,true><<<g, blk, 0, stream>>>(qh, (long)H_ * N_ * MF, (long)N_ * MF, MF,
                                                       ctx, (long)H_ * MF * DH_, (long)MF * DH_, DH_,
                                                       Ob, (long)N_ * DM, DH_, DM,
                                                       nullptr, dnv, (long)H_ * N_, N_,
                                                       MF, 1.0f, 0, H_);
  }
  // effective attention map (dominant GEMM, async f16 staging)
  {
    dim3 g(N_ / 64, N_ / 128, B_);
    attn_wmma<<<g, blk, 0, stream>>>(qh, kh, dnv, attnOut);
  }
  // O1 = out @ Wo + bo ; h1 = LN1
  {
    dim3 g(DM / 64, TOK / 128, 1);
    gemm_wmma<false,false,false><<<g, blk, 0, stream>>>(Ob, 0, 0, DM, Wo, 0, 0, DM, O1, 0, 0, DM,
                                                        bo, nullptr, 0, 0, DM, 1.0f, 0, 1);
  }
  ln_kernel<<<(int)TOK, blk, 0, stream>>>(O1, g1, be1, H1b);
  // FF = relu(h1 @ W1 + b1)
  {
    dim3 g(DFF / 64, TOK / 128, 1);
    gemm_wmma<false,false,false><<<g, blk, 0, stream>>>(H1b, 0, 0, DM, W1, 0, 0, DFF, FFb, 0, 0, DFF,
                                                        b1, nullptr, 0, 0, DM, 1.0f, 1, 1);
  }
  // O2 = FF @ W2 + b2 ; y = LN2
  {
    dim3 g(DM / 64, TOK / 128, 1);
    gemm_wmma<false,false,false><<<g, blk, 0, stream>>>(FFb, 0, 0, DFF, W2, 0, 0, DM, O2, 0, 0, DM,
                                                        b2, nullptr, 0, 0, DM, 1.0f, 0, 1);
  }
  ln_kernel<<<(int)TOK, blk, 0, stream>>>(O2, g2, be2, yOut);
}
